// SelfMultiheadAttn_53566832115874
// MI455X (gfx1250) — matrix operand order repaired and checked by reading, not compiled, not run on hardware
//
#include <hip/hip_runtime.h>

// ---------------- problem constants ----------------
#define T_DIM 2048
#define B_DIM 4
#define E_DIM 1024
#define H_DIM 16
#define D_DIM 64
#define M_ROWS (T_DIM * B_DIM)   // 8192
#define N_QKV  (3 * E_DIM)       // 3072
#define BH     (B_DIM * H_DIM)   // 64

// ---------------- WMMA plumbing ----------------
typedef __attribute__((ext_vector_type(16))) __bf16 bf16x16;
typedef __attribute__((ext_vector_type(8)))  float  f32x8;

union Frag { bf16x16 v; unsigned int u[8]; };

__device__ __forceinline__ unsigned short f2bf(float f) {
  unsigned int u = __float_as_uint(f);
  u += 0x7FFFu + ((u >> 16) & 1u);   // round-to-nearest-even
  return (unsigned short)(u >> 16);
}

__device__ __forceinline__ f32x8 wmma_bf16(const Frag& a, const Frag& b, f32x8 c) {
  // D = A(16x32 bf16) * B(32x16 bf16) + C(16x16 f32)
  return __builtin_amdgcn_wmma_f32_16x16x32_bf16(false, a.v, false, b.v,
                                                 (short)0, c, false, false);
}

// Fragment loads from bf16 row-major, K-contiguous (NT) operands.
// A layout (ISA 16-bit 16x32): dword p -> col k0 + 8*half + 2p (+8 if p>=4)
__device__ __forceinline__ void load_afrag(Frag& a, const unsigned int* arow,
                                           int k0, int half) {
#pragma unroll
  for (int p = 0; p < 8; ++p) {
    int ca = k0 + 8 * half + 2 * p + ((p >= 4) ? 8 : 0);
    a.u[p] = arow[ca >> 1];
  }
}
// B layout: dword p -> col k0 + 8*half + 2p (+8 if p>=4)
__device__ __forceinline__ void load_bfrag(Frag& b, const unsigned int* brow,
                                           int k0, int half) {
#pragma unroll
  for (int p = 0; p < 8; ++p) {
    int cb = k0 + 8 * half + 2 * p + ((p >= 4) ? 8 : 0);
    b.u[p] = brow[cb >> 1];
  }
}

// ---------------- fp32 -> bf16 convert ----------------
__global__ void f32_to_bf16_kernel(const float* __restrict__ src,
                                   unsigned short* __restrict__ dst, int n) {
  int i = blockIdx.x * blockDim.x + threadIdx.x;
  if (i < n) dst[i] = f2bf(src[i]);
}

// ---------------- NT WMMA GEMM: C[M,N] = A[M,K] * B[N,K]^T + bias[N]
// Wave tile 32x64 (2 A-frags x 4 B-frags, 8 accumulators).
// Block 128 = 4 waves arranged 2x2 -> block tile 64x128.
// grid = (N/128, M/64)
__global__ void __launch_bounds__(128)
gemm_nt_bf16(const unsigned short* __restrict__ A,
             const unsigned short* __restrict__ Bm,
             const float* __restrict__ bias,
             float* __restrict__ C,
             int M, int N, int K) {
  int lane = threadIdx.x & 31;
  int wave = threadIdx.x >> 5;      // 0..3
  int half = lane >> 4;
  int ln   = lane & 15;
  int m0 = blockIdx.y * 64 + (wave >> 1) * 32;
  int n0 = blockIdx.x * 128 + (wave & 1) * 64;

  const unsigned int* arow0 = (const unsigned int*)(A + (size_t)(m0 + ln) * K);
  const unsigned int* arow1 = (const unsigned int*)(A + (size_t)(m0 + 16 + ln) * K);
  const unsigned int* brow[4];
#pragma unroll
  for (int j = 0; j < 4; ++j)
    brow[j] = (const unsigned int*)(Bm + (size_t)(n0 + 16 * j + ln) * K);

  f32x8 acc[8];
#pragma unroll
  for (int i = 0; i < 8; ++i) acc[i] = (f32x8){0.f,0.f,0.f,0.f,0.f,0.f,0.f,0.f};

  for (int k0 = 0; k0 < K; k0 += 32) {
    // speculative prefetch one k-step ahead (global_prefetch_b8; OOB dropped)
    __builtin_prefetch((const char*)arow0 + (size_t)(k0 + 32) * 2, 0, 1);
    __builtin_prefetch((const char*)arow1 + (size_t)(k0 + 32) * 2, 0, 1);
#pragma unroll
    for (int j = 0; j < 4; ++j)
      __builtin_prefetch((const char*)brow[j] + (size_t)(k0 + 32) * 2, 0, 1);

    Frag a0, a1;
    load_afrag(a0, arow0, k0, half);
    load_afrag(a1, arow1, k0, half);
#pragma unroll
    for (int j = 0; j < 4; ++j) {
      Frag b;
      load_bfrag(b, brow[j], k0, half);
      acc[2 * j]     = wmma_bf16(a0, b, acc[2 * j]);
      acc[2 * j + 1] = wmma_bf16(a1, b, acc[2 * j + 1]);
    }
  }
#pragma unroll
  for (int j = 0; j < 4; ++j) {
    int col = n0 + 16 * j + ln;
    float bz = bias[col];
#pragma unroll
    for (int v = 0; v < 8; ++v) {
      int r0 = m0 + v + 8 * half;         // C: row = vgpr + 8*half
      C[(size_t)r0 * N + col]        = acc[2 * j][v] + bz;
      C[(size_t)(r0 + 16) * N + col] = acc[2 * j + 1][v] + bz;
    }
  }
}

// ---------------- RoPE + scale + repack ----------------
// qkv fp32 [8192, 3072] -> qb/kb bf16 [BH, T, D], vT bf16 [BH, D, T]
__global__ void rope_pack_kernel(const float* __restrict__ qkv,
                                 const float* __restrict__ pos,
                                 unsigned short* __restrict__ qb,
                                 unsigned short* __restrict__ kb,
                                 unsigned short* __restrict__ vT) {
  int idx = blockIdx.x * blockDim.x + threadIdx.x;
  if (idx >= M_ROWS * E_DIM) return;
  int d   = idx & (D_DIM - 1);
  int h   = (idx >> 6) & (H_DIM - 1);
  int row = idx >> 10;            // t*B + b
  int b   = row & (B_DIM - 1);
  int t   = row >> 2;
  const float* qr = qkv + (size_t)row * N_QKV;
  int col = h * D_DIM + d;
  float q = qr[col];
  float k = qr[E_DIM + col];
  float v = qr[2 * E_DIM + col];
  float ang = pos[t * D_DIM + d];
  float c = __cosf(ang), s = __sinf(ang);
  int   dp  = (d < 32) ? d + 32 : d - 32;
  float sgn = (d < 32) ? -1.f : 1.f;
  float qo = qr[h * D_DIM + dp];
  float ko = qr[E_DIM + h * D_DIM + dp];
  float qn = (q * c + sgn * qo * s) * 0.125f;   // scaling = 64^-0.5
  float kn = (k * c + sgn * ko * s);
  int bh = b * H_DIM + h;
  qb[((size_t)bh * T_DIM + t) * D_DIM + d] = f2bf(qn);
  kb[((size_t)bh * T_DIM + t) * D_DIM + d] = f2bf(kn);
  vT[((size_t)bh * D_DIM + d) * T_DIM + t] = f2bf(v);
}

// ---------------- fused: scores + causal softmax + ctx = P @ V ----------------
// grid = (T/16, BH), block = 256 (8 waves)
// LDS: 16 prob rows of SROW floats + 4*256 floats reduction scratch = 136192 B
#define SROW 2064
#define RED_OFF (16 * SROW)
__global__ void __launch_bounds__(256)
attn_fused_kernel(const unsigned short* __restrict__ qb,
                  const unsigned short* __restrict__ kb,
                  const unsigned short* __restrict__ vT,
                  float* __restrict__ attn,
                  unsigned short* __restrict__ ctx) {
  extern __shared__ float sc[];   // [16][SROW] + [4][256]
  int t0 = blockIdx.x * 16;
  int bh = blockIdx.y;
  int lane = threadIdx.x & 31;
  int wave = threadIdx.x >> 5;
  int half = lane >> 4;
  int ln   = lane & 15;

  // ---- phase 1: scores via WMMA (q A-frags reused across all column tiles)
  Frag a0, a1;
  const unsigned int* qrow =
      (const unsigned int*)(qb + ((size_t)bh * T_DIM + t0 + ln) * D_DIM);
  load_afrag(a0, qrow, 0, half);
  load_afrag(a1, qrow, 32, half);

  int tmax = t0 + 15;
  for (int tile = wave; tile < T_DIM / 16; tile += 8) {
    int s0 = tile * 16;
    if (s0 > tmax) continue;      // fully-masked tile: never read later
    Frag b0, b1;
    const unsigned int* krow =
        (const unsigned int*)(kb + ((size_t)bh * T_DIM + s0 + ln) * D_DIM);
    load_bfrag(b0, krow, 0, half);
    load_bfrag(b1, krow, 32, half);
    f32x8 acc = {0.f, 0.f, 0.f, 0.f, 0.f, 0.f, 0.f, 0.f};
    acc = wmma_bf16(a0, b0, acc);
    acc = wmma_bf16(a1, b1, acc);
#pragma unroll
    for (int v = 0; v < 8; ++v) {
      int r = v + 8 * half;
      int t = t0 + r;
      int s = s0 + ln;
      sc[r * SROW + s0 + ln] = (s <= t) ? acc[v] : -__builtin_inff();
    }
  }
  __syncthreads();

  // ---- phase 2: softmax (float4-vectorized row passes);
  // overwrite score rows with probabilities, zero-padded to s < t0+32
  // so the phase-3 k-loop reads exact zeros beyond the causal frontier.
  int fill_end = t0 + 32;                     // <= SROW
  for (int rr = 0; rr < 2; ++rr) {
    int r = wave + rr * 8;
    int t = t0 + r;
    float* srow = sc + (size_t)r * SROW;
    const float NEG_INF = -__builtin_inff();
    // pass 1: row max (per-element causal select; garbage beyond frontier masked)
    float mx = NEG_INF;
    for (int s = lane * 4; s < fill_end; s += 128) {
      float4 x = *(const float4*)&srow[s];
      mx = fmaxf(mx, (s + 0 <= t) ? x.x : NEG_INF);
      mx = fmaxf(mx, (s + 1 <= t) ? x.y : NEG_INF);
      mx = fmaxf(mx, (s + 2 <= t) ? x.z : NEG_INF);
      mx = fmaxf(mx, (s + 3 <= t) ? x.w : NEG_INF);
    }
#pragma unroll
    for (int off = 16; off; off >>= 1) mx = fmaxf(mx, __shfl_xor(mx, off, 32));
    // pass 2: e = exp(x - mx) stored in place (zeros where masked), row sum
    float sum = 0.f;
    for (int s = lane * 4; s < fill_end; s += 128) {
      float4 x = *(const float4*)&srow[s];
      float4 e;
      e.x = (s + 0 <= t) ? __expf(x.x - mx) : 0.f;
      e.y = (s + 1 <= t) ? __expf(x.y - mx) : 0.f;
      e.z = (s + 2 <= t) ? __expf(x.z - mx) : 0.f;
      e.w = (s + 3 <= t) ? __expf(x.w - mx) : 0.f;
      *(float4*)&srow[s] = e;
      sum += e.x + e.y + e.z + e.w;
    }
#pragma unroll
    for (int off = 16; off; off >>= 1) sum += __shfl_xor(sum, off, 32);
    float inv = 1.0f / sum;
    // pass 3: normalize in LDS + b128 stores of attn row; then zero tail
    float* arow = attn + ((size_t)bh * T_DIM + t) * T_DIM;
    for (int s = lane * 4; s < fill_end; s += 128) {
      float4 e = *(const float4*)&srow[s];
      float4 p = {e.x * inv, e.y * inv, e.z * inv, e.w * inv};
      *(float4*)&srow[s] = p;
      if (s < T_DIM) *(float4*)&arow[s] = p;
    }
    float4 z4 = {0.f, 0.f, 0.f, 0.f};
    for (int s = fill_end + lane * 4; s < T_DIM; s += 128)
      *(float4*)&arow[s] = z4;
  }
  __syncthreads();

  // ---- phase 3: ctx(16 x 64) = P(16 x kTot) @ V(kTot x 64)
  // 8 waves = 4 d-tiles x 2 interleaved k-halves; A-frags from LDS probs.
  int j  = wave & 3;              // d-tile
  int kh = wave >> 2;             // k interleave phase
  int b = bh >> 4, h = bh & 15;
  const unsigned int* vrow =
      (const unsigned int*)(vT + ((size_t)bh * D_DIM + 16 * j + ln) * T_DIM);
  f32x8 acc = {0.f, 0.f, 0.f, 0.f, 0.f, 0.f, 0.f, 0.f};
  int kTot = t0 + 16;             // causal extent (P is 0 beyond, up to t0+32)
  for (int k0 = kh * 32; k0 < kTot; k0 += 64) {
    Frag a;
#pragma unroll
    for (int p = 0; p < 8; ++p) {
      int ca = k0 + 8 * half + 2 * p + ((p >= 4) ? 8 : 0);
      float2 x = *(const float2*)&sc[ln * SROW + ca];   // 8B-aligned (SROW even)
      a.u[p] = (unsigned int)f2bf(x.x) | ((unsigned int)f2bf(x.y) << 16);
    }
    Frag bf;
    load_bfrag(bf, vrow, k0, half);
    acc = wmma_bf16(a, bf, acc);
  }
  // combine the two k-halves: waves 4..7 park partials in LDS scratch
  if (wave >= 4) {
    float* red = sc + RED_OFF + (wave - 4) * 256;
#pragma unroll
    for (int v = 0; v < 8; ++v) red[v * 32 + lane] = acc[v];
  }
  __syncthreads();
  if (wave < 4) {
    const float* red = sc + RED_OFF + wave * 256;
#pragma unroll
    for (int v = 0; v < 8; ++v) {
      int t = t0 + v + 8 * half;
      float r = acc[v] + red[v * 32 + lane];
      ctx[((size_t)t * B_DIM + b) * E_DIM + h * D_DIM + 16 * j + ln] = f2bf(r);
    }
  }
}

// ---------------- launcher ----------------
extern "C" void kernel_launch(void* const* d_in, const int* in_sizes, int n_in,
                              void* d_out, int out_size, void* d_ws, size_t ws_size,
                              hipStream_t stream) {
  (void)in_sizes; (void)n_in; (void)out_size; (void)ws_size;
  const float* inp   = (const float*)d_in[0];   // [T,B,E]
  const float* pos   = (const float*)d_in[1];   // [T,D]
  const float* in_w  = (const float*)d_in[2];   // [3E,E]
  const float* in_b  = (const float*)d_in[3];   // [3E]
  const float* out_w = (const float*)d_in[4];   // [E,E]
  const float* out_b = (const float*)d_in[5];   // [E]
  // d_in[6] = attn_mask: causal, computed analytically

  float* out_ptr  = (float*)d_out;                                // [T,B,E]
  float* attn_ptr = out_ptr + (size_t)T_DIM * B_DIM * E_DIM;      // [B,H,T,T]

  char* ws = (char*)d_ws;
  const size_t XB_OFF  = 0;                                           // 16 MB
  const size_t WB_OFF  = XB_OFF  + (size_t)M_ROWS * E_DIM * 2;        // 6 MB
  const size_t OWB_OFF = WB_OFF  + (size_t)N_QKV * E_DIM * 2;         // 2 MB
  const size_t QKV_OFF = OWB_OFF + (size_t)E_DIM * E_DIM * 2;         // 96 MB fp32
  const size_t QB_OFF  = QKV_OFF + (size_t)M_ROWS * N_QKV * 4;        // 16 MB
  const size_t KB_OFF  = QB_OFF  + (size_t)BH * T_DIM * D_DIM * 2;    // 16 MB
  const size_t VT_OFF  = KB_OFF  + (size_t)BH * T_DIM * D_DIM * 2;    // 16 MB
  const size_t CTX_OFF = QKV_OFF;  // reuse qkv region after it is consumed

  unsigned short* xb   = (unsigned short*)(ws + XB_OFF);
  unsigned short* wb   = (unsigned short*)(ws + WB_OFF);
  unsigned short* owb  = (unsigned short*)(ws + OWB_OFF);
  float*          qkv  = (float*)(ws + QKV_OFF);
  unsigned short* qb   = (unsigned short*)(ws + QB_OFF);
  unsigned short* kb   = (unsigned short*)(ws + KB_OFF);
  unsigned short* vT   = (unsigned short*)(ws + VT_OFF);
  unsigned short* ctxb = (unsigned short*)(ws + CTX_OFF);

  // 1) converts
  {
    int n = M_ROWS * E_DIM;
    f32_to_bf16_kernel<<<(n + 255) / 256, 256, 0, stream>>>(inp, xb, n);
    n = N_QKV * E_DIM;
    f32_to_bf16_kernel<<<(n + 255) / 256, 256, 0, stream>>>(in_w, wb, n);
    n = E_DIM * E_DIM;
    f32_to_bf16_kernel<<<(n + 255) / 256, 256, 0, stream>>>(out_w, owb, n);
  }
  // 2) fused QKV projection + bias  (grid: N/128 x M/64)
  gemm_nt_bf16<<<dim3(N_QKV / 128, M_ROWS / 64), 128, 0, stream>>>(
      xb, wb, in_b, qkv, M_ROWS, N_QKV, E_DIM);
  // 3) RoPE + scaling + repack
  {
    int n = M_ROWS * E_DIM;
    rope_pack_kernel<<<(n + 255) / 256, 256, 0, stream>>>(qkv, pos, qb, kb, vT);
  }
  // 4) fused scores + causal softmax + attn store + ctx = P @ V
  attn_fused_kernel<<<dim3(T_DIM / 16, BH), 256, (RED_OFF + 4 * 256) * 4,
                      stream>>>(qb, kb, vT, attn_ptr, ctxb);
  // 5) output projection + bias
  gemm_nt_bf16<<<dim3(E_DIM / 128, M_ROWS / 64), 128, 0, stream>>>(
      ctxb, owb, out_b, out_ptr, M_ROWS, E_DIM, E_DIM);
}
